// FujiRoutedExperts_2611340116636
// MI455X (gfx1250) — matrix operand order repaired and checked by reading, not compiled, not run on hardware
//
#include <hip/hip_runtime.h>
#include <stdint.h>

// Problem constants (match reference)
#define E_EXP 32
#define D_DIM 1024
#define I_DIM 512
#define T_TOK 4096
#define K_TOP 8
#define NSLOT (T_TOK * K_TOP)        // 32768 routed slots
#define M_TILE 32                    // gathered tokens per block
#define MAX_TILES (NSLOT / M_TILE)   // 1024 (worst case: one expert gets everything)

typedef __attribute__((ext_vector_type(16))) __bf16          v16bf;
typedef __attribute__((ext_vector_type(8)))  float           v8f;
typedef __attribute__((ext_vector_type(8)))  unsigned short  v8u;

union Frag {
  v8u   h[2];
  v16bf bf;
};

__device__ __forceinline__ unsigned short f2bf(float f) {
  union { float f; uint32_t u; } x; x.f = f;
  uint32_t u = x.u;
  uint32_t r = u + 0x7FFFu + ((u >> 16) & 1u);  // round-to-nearest-even
  return (unsigned short)(r >> 16);
}

// ---------------------------------------------------------------- fp32 -> bf16
__global__ void cvt_f32_bf16(const float* __restrict__ src,
                             unsigned short* __restrict__ dst, int n8) {
  int i = blockIdx.x * blockDim.x + threadIdx.x;
  int stride = gridDim.x * blockDim.x;
  for (; i < n8; i += stride) {
    const float4* s = (const float4*)src + (size_t)i * 2;
    float4 a = s[0], b = s[1];
    v8u o;
    o[0] = f2bf(a.x); o[1] = f2bf(a.y); o[2] = f2bf(a.z); o[3] = f2bf(a.w);
    o[4] = f2bf(b.x); o[5] = f2bf(b.y); o[6] = f2bf(b.z); o[7] = f2bf(b.w);
    *((v8u*)dst + i) = o;
  }
}

// ---------------------------------------------------------------- zeroing
__global__ void zero_f32(float* __restrict__ p, int n4) {
  int i = blockIdx.x * blockDim.x + threadIdx.x;
  int stride = gridDim.x * blockDim.x;
  float4 z = {0.f, 0.f, 0.f, 0.f};
  for (; i < n4; i += stride) ((float4*)p)[i] = z;
}

__global__ void zero_meta(int* __restrict__ meta) {
  if (threadIdx.x < 64) meta[threadIdx.x] = 0;   // counts[32] + cursor[32]
}

// ---------------------------------------------------------------- routing
__global__ void route_count(const int* __restrict__ idx, int* __restrict__ counts) {
  int i = blockIdx.x * blockDim.x + threadIdx.x;
  if (i < NSLOT) atomicAdd(&counts[idx[i] & (E_EXP - 1)], 1);
}

__global__ void route_scan(const int* __restrict__ counts, int* __restrict__ offs,
                           int* __restrict__ cursor) {
  if (threadIdx.x == 0) {
    int s = 0;
    for (int e = 0; e < E_EXP; ++e) { offs[e] = s; s += counts[e]; cursor[e] = 0; }
    offs[E_EXP] = s;
  }
}

__global__ void route_scatter(const int* __restrict__ idx, const float* __restrict__ w,
                              const int* __restrict__ offs, int* __restrict__ cursor,
                              int* __restrict__ tokIds, float* __restrict__ slotW) {
  int i = blockIdx.x * blockDim.x + threadIdx.x;
  if (i < NSLOT) {
    int e = idx[i] & (E_EXP - 1);
    int pos = offs[e] + atomicAdd(&cursor[e], 1);
    tokIds[pos] = i >> 3;          // token = slot / K
    slotW[pos]  = w[i];
  }
}

// ---------------------------------------------------------------- fused expert GEMM
// Block: 256 threads = 8 wave32s. One (expert, 32-token tile).
// Phase 1: h = silu(x @ Wg^T) * (x @ Wu^T)  -> bf16 tile in LDS (32 x 512)
// Phase 2: y = h @ Wd^T, scaled by combine weight, atomic-add into out.
__global__ void __launch_bounds__(256)
moe_expert_gemm(const unsigned short* __restrict__ hidB,
                const unsigned short* __restrict__ guB,
                const unsigned short* __restrict__ dnB,
                const int* __restrict__ counts,
                const int* __restrict__ offs,
                const int* __restrict__ tokIds,
                const float* __restrict__ slotW,
                float* __restrict__ out) {
  const int e    = blockIdx.y;
  const int tile = blockIdx.x;
  const int cnt  = counts[e];
  if (tile * M_TILE >= cnt) return;
  const int base = offs[e] + tile * M_TILE;

  __shared__ int            sTok[M_TILE];
  __shared__ float          sW[M_TILE];
  __shared__ unsigned short sH[M_TILE * I_DIM];   // 32 KB

  const int tid = threadIdx.x;
  if (tid < M_TILE) {
    bool valid = (tile * M_TILE + tid) < cnt;
    sTok[tid] = valid ? tokIds[base + tid] : 0;
    sW[tid]   = valid ? slotW[base + tid]  : 0.f;
  }
  __syncthreads();

  const int lane   = tid & 31;
  const int wid    = tid >> 5;
  const int lm     = lane & 15;       // output row/col within 16-tile
  const int hiHalf = lane >> 4;       // 0 | 1
  const int aSel   = hiHalf * 16;     // A frag K-offset select (contiguous K, matches B)
  const int bSel   = hiHalf * 16;     // B frag K-offset select
  const int msub   = wid >> 2;        // which 16-row subtile (0..1)
  const int colQ   = wid & 3;         // column quarter
  const int rowBlk = msub * 16;

  const int myTok = sTok[rowBlk + lm];
  const unsigned short* aRow = hidB + (size_t)myTok * D_DIM;

  // ---------------- Phase 1: gate/up GEMM + SwiGLU -> LDS ----------------
  for (int jt = 0; jt < 8; ++jt) {
    const int nTile = (colQ * 8 + jt) * 16;                       // 0..511
    const unsigned short* bgRow =
        guB + ((size_t)e * (2 * I_DIM) + nTile + lm) * D_DIM;
    const unsigned short* buRow =
        guB + ((size_t)e * (2 * I_DIM) + I_DIM + nTile + lm) * D_DIM;
    __builtin_prefetch(bgRow + 16 * D_DIM, 0, 1);
    __builtin_prefetch(buRow + 16 * D_DIM, 0, 1);

    v8f accG = {}; v8f accU = {};
    for (int kb = 0; kb < D_DIM; kb += 32) {
      Frag a, bg, bu;
      a.h[0]  = *(const v8u*)(aRow  + kb + aSel);
      a.h[1]  = *(const v8u*)(aRow  + kb + aSel + 8);
      bg.h[0] = *(const v8u*)(bgRow + kb + bSel);
      bg.h[1] = *(const v8u*)(bgRow + kb + bSel + 8);
      bu.h[0] = *(const v8u*)(buRow + kb + bSel);
      bu.h[1] = *(const v8u*)(buRow + kb + bSel + 8);
      accG = __builtin_amdgcn_wmma_f32_16x16x32_bf16(false, a.bf, false, bg.bf,
                                                     (short)0, accG, false, false);
      accU = __builtin_amdgcn_wmma_f32_16x16x32_bf16(false, a.bf, false, bu.bf,
                                                     (short)0, accU, false, false);
    }
    const int col = nTile + lm;
#pragma unroll
    for (int r = 0; r < 8; ++r) {
      float g = accG[r], u = accU[r];
      float h = (g / (1.f + __expf(-g))) * u;     // SwiGLU
      sH[(rowBlk + r + hiHalf * 8) * I_DIM + col] = f2bf(h);
    }
  }
  __syncthreads();

  // ---------------- Phase 2: down projection + weighted scatter ----------------
  const unsigned short* aRow2 = sH + (size_t)(rowBlk + lm) * I_DIM;
  for (int jt = 0; jt < 16; ++jt) {
    const int nTile = (colQ * 16 + jt) * 16;                      // 0..1023
    const unsigned short* bRow =
        dnB + ((size_t)e * D_DIM + nTile + lm) * I_DIM;
    __builtin_prefetch(bRow + 16 * I_DIM, 0, 1);

    v8f acc = {};
    for (int kb = 0; kb < I_DIM; kb += 32) {
      Frag a, b;
      a.h[0] = *(const v8u*)(aRow2 + kb + aSel);          // ds_load_b128 from LDS
      a.h[1] = *(const v8u*)(aRow2 + kb + aSel + 8);
      b.h[0] = *(const v8u*)(bRow  + kb + bSel);
      b.h[1] = *(const v8u*)(bRow  + kb + bSel + 8);
      acc = __builtin_amdgcn_wmma_f32_16x16x32_bf16(false, a.bf, false, b.bf,
                                                    (short)0, acc, false, false);
    }
    const int d = nTile + lm;
#pragma unroll
    for (int r = 0; r < 8; ++r) {
      const int m = rowBlk + r + hiHalf * 8;
      float val = sW[m] * acc[r];
      atomicAdd(out + (size_t)sTok[m] * D_DIM + d, val);  // padded rows add 0
    }
  }
}

// ---------------------------------------------------------------- host entry
extern "C" void kernel_launch(void* const* d_in, const int* in_sizes, int n_in,
                              void* d_out, int out_size, void* d_ws, size_t ws_size,
                              hipStream_t stream) {
  const float* hidden = (const float*)d_in[0];   // [T, D]
  const int*   topIdx = (const int*)  d_in[1];   // [T, K]
  const float* topW   = (const float*)d_in[2];   // [T, K]
  const float* gu     = (const float*)d_in[3];   // [E, 2I, D]
  const float* dn     = (const float*)d_in[4];   // [E, D, I]
  float* out = (float*)d_out;
  char*  ws  = (char*)d_ws;

  // Workspace layout (~104.6 MB total)
  constexpr size_t HID_OFF  = 0;
  constexpr size_t GU_OFF   = HID_OFF + (size_t)T_TOK * D_DIM * 2;               // +8 MB
  constexpr size_t DN_OFF   = GU_OFF  + (size_t)E_EXP * 2 * I_DIM * D_DIM * 2;   // +64 MB
  constexpr size_t META_OFF = DN_OFF  + (size_t)E_EXP * D_DIM * I_DIM * 2;       // +32 MB

  unsigned short* hidB = (unsigned short*)(ws + HID_OFF);
  unsigned short* guB  = (unsigned short*)(ws + GU_OFF);
  unsigned short* dnB  = (unsigned short*)(ws + DN_OFF);
  int* meta    = (int*)(ws + META_OFF);
  int* counts  = meta;            // [32]
  int* cursor  = meta + 32;       // [32]
  int* offs    = meta + 64;       // [33]
  int* tokIds  = meta + 128;      // [32768]
  float* slotW = (float*)(meta + 128 + NSLOT);   // [32768]

  // 1) Pre-convert fp32 -> bf16 (one HBM pass; bf16 weights fit in 192 MB L2)
  cvt_f32_bf16<<<2048, 256, 0, stream>>>(hidden, hidB, (T_TOK * D_DIM) / 8);
  cvt_f32_bf16<<<16384, 256, 0, stream>>>(gu, guB, (E_EXP * 2 * I_DIM * D_DIM) / 8);
  cvt_f32_bf16<<<8192, 256, 0, stream>>>(dn, dnB, (E_EXP * D_DIM * I_DIM) / 8);

  // 2) Zero output + routing meta
  zero_f32<<<4096, 256, 0, stream>>>(out, (T_TOK * D_DIM) / 4);
  zero_meta<<<1, 64, 0, stream>>>(meta);

  // 3) Routing: histogram -> scan -> scatter into per-expert segments
  route_count<<<NSLOT / 256, 256, 0, stream>>>(topIdx, counts);
  route_scan<<<1, 32, 0, stream>>>(counts, offs, cursor);
  route_scatter<<<NSLOT / 256, 256, 0, stream>>>(topIdx, topW, offs, cursor,
                                                 tokIds, slotW);

  // 4) Fused expert GEMM (bf16 WMMA), grid = (max tiles, experts); empty tiles exit
  dim3 grid(MAX_TILES, E_EXP);
  moe_expert_gemm<<<grid, 256, 0, stream>>>(hidB, guB, dnB, counts, offs,
                                            tokIds, slotW, out);
}
